// AttentionTwoStream_31903017075408
// MI455X (gfx1250) — compile-verified, matrix operand order checked by reading, not run on hardware
//
#include <hip/hip_runtime.h>
#include <hip/hip_bf16.h>

#define HDIM 512
#define BDIM 64
#define TV   512
#define TT   64
#define BPAD 520   // LDS row stride (elements) for B tiles: 512 + 8 to spread banks

typedef __attribute__((ext_vector_type(16))) __bf16 bf16x16;
typedef __attribute__((ext_vector_type(8)))  __bf16 bf16x8;
typedef __attribute__((ext_vector_type(8)))  float  f32x8;

__device__ __forceinline__ __bf16 f32_to_bf16(float f) {
    unsigned int u = __float_as_uint(f);
    u = (u + 0x7FFFu + ((u >> 16) & 1u)) >> 16;   // round-to-nearest-even
    unsigned short s = (unsigned short)u;
    return __builtin_bit_cast(__bf16, s);
}

// Load one WMMA operand fragment for K-step k0 from a row pointer (stride HDIM).
// Per ISA 7.12.2 (16-bit 16x32 operand): lanes 0-15 hold K {k0..k0+7, k0+16..k0+23},
// lanes 16-31 hold K {k0+8..k0+15, k0+24..k0+31}. Two 16-byte loads per lane.
__device__ __forceinline__ bf16x16 load_op(const __bf16* row, int k0, int sel) {
    union { bf16x16 v; bf16x8 h[2]; } u;
    u.h[0] = *reinterpret_cast<const bf16x8*>(row + k0 + sel);
    u.h[1] = *reinterpret_cast<const bf16x8*>(row + k0 + 16 + sel);
    return u.v;
}

// Same fragment layout, but reading from an LDS-resident tile row (stride BPAD).
__device__ __forceinline__ bf16x16 load_op_lds(const __bf16* row, int k0, int sel) {
    union { bf16x16 v; bf16x8 h[2]; } u;
    u.h[0] = *reinterpret_cast<const bf16x8*>(row + k0 + sel);
    u.h[1] = *reinterpret_cast<const bf16x8*>(row + k0 + 16 + sel);
    return u.v;
}

// Async-copy one 16x512 bf16 B tile (rows n_base..n_base+15 of Bt[512,512]) into LDS.
// 1024 16-byte chunks, 256 threads -> 4 async b128 transfers per thread (ASYNCcnt).
__device__ __forceinline__ void async_b_tile(const __bf16* __restrict__ Bt, int n_base,
                                             __bf16* dst /* LDS, rows stride BPAD */) {
#pragma unroll
    for (int i = 0; i < 4; ++i) {
        const int c   = threadIdx.x + i * 256;   // chunk id 0..1023
        const int row = c >> 6;                  // 64 chunks per row
        const int col = (c & 63) << 3;           // element column, multiple of 8
        const __bf16* g = Bt + (size_t)(n_base + row) * HDIM + col;
        const unsigned int l = (unsigned int)(uintptr_t)(dst + row * BPAD + col);
        asm volatile("global_load_async_to_lds_b128 %0, %1, off"
                     :: "v"(l), "v"(g) : "memory");
    }
}

__device__ __forceinline__ void wait_async0() {
    asm volatile("s_wait_asynccnt 0x0" ::: "memory");
}

// ---------------- conversion kernels ----------------
__global__ void cvt_bf16(const float* __restrict__ in, __bf16* __restrict__ out, int n) {
    int i = blockIdx.x * blockDim.x + threadIdx.x;
    if (i < n) out[i] = f32_to_bf16(in[i]);
}

// Wt[j*H + k] = bf16(W[k*H + j])   (produce [N,K] operand for X @ W)
__global__ void cvt_tr_bf16(const float* __restrict__ W, __bf16* __restrict__ Wt) {
    __shared__ float tile[16][17];
    int x = blockIdx.x * 16 + threadIdx.x;
    int y = blockIdx.y * 16 + threadIdx.y;
    tile[threadIdx.y][threadIdx.x] = W[(size_t)y * HDIM + x];
    __syncthreads();
    int ox = blockIdx.y * 16 + threadIdx.x;
    int oy = blockIdx.x * 16 + threadIdx.y;
    Wt[(size_t)oy * HDIM + ox] = f32_to_bf16(tile[threadIdx.x][threadIdx.y]);
}

// ---------------- generic WMMA GEMM (small M): C = A @ Bt^T (+bias, +addMat) ----------
__global__ __launch_bounds__(256, 1)
void gemm_wmma(const __bf16* __restrict__ A, const __bf16* __restrict__ Bt,
               float* __restrict__ C,
               const float* __restrict__ biasN,   // [H] or null
               const float* __restrict__ addMat,  // [M,H] or null
               int M) {
    const int lane   = threadIdx.x & 31;
    const int wave   = threadIdx.x >> 5;
    const int mtile  = blockIdx.x * 8 + wave;
    const int m_base = mtile * 16;
    if (m_base >= M) return;                       // wave-uniform; no barriers below
    const int laneHi = lane >> 4;
    const int ln     = lane & 15;
    const int sel    = laneHi ? 8 : 0;

    // A strip (16 rows x 512 K) resident in registers: 16 x v16bf = 128 VGPRs.
    const __bf16* arow = A + (size_t)(m_base + ln) * HDIM;
    bf16x16 a[16];
#pragma unroll
    for (int kk = 0; kk < 16; ++kk) a[kk] = load_op(arow, kk * 32, sel);

    for (int n_base = 0; n_base < HDIM; n_base += 16) {
        const __bf16* brow = Bt + (size_t)(n_base + ln) * HDIM;
        f32x8 acc = {};
#pragma unroll
        for (int kk = 0; kk < 16; ++kk) {
            bf16x16 b = load_op(brow, kk * 32, sel);
            acc = __builtin_amdgcn_wmma_f32_16x16x32_bf16(false, a[kk], false, b,
                                                          (short)0, acc, false, false);
        }
        const int n = n_base + ln;
        const float bn = biasN ? biasN[n] : 0.0f;
        union { f32x8 v; float f[8]; } cu; cu.v = acc;
#pragma unroll
        for (int r = 0; r < 8; ++r) {
            const int m = m_base + r + laneHi * 8;
            float v = cu.f[r] + bn;
            if (addMat) v += addMat[(size_t)m * HDIM + n];
            C[(size_t)m * HDIM + n] = v;
        }
    }
}

// ---------- fused score GEMM: score[m] = sum_n tanh((A@W)[m,n] + Uh[b,n] + bias[n]) * V[n]
// A strip register-resident; B tiles double-buffered in LDS via async-to-LDS copies.
// Grid is exact: M is a multiple of 128 (8 waves/block), so no early-outs vs barriers.
__global__ __launch_bounds__(256, 1)
void gemm_score(const __bf16* __restrict__ A, const __bf16* __restrict__ Bt,
                const float* __restrict__ Uh,    // [B,H]
                const float* __restrict__ biasN, // [H]
                const float* __restrict__ Vv,    // [H]
                float* __restrict__ score,       // [M]
                int T) {
    __shared__ __align__(16) __bf16 sB[2][16 * BPAD];   // 2 x 16.6 KB

    const int lane   = threadIdx.x & 31;
    const int wave   = threadIdx.x >> 5;
    const int m_base = (blockIdx.x * 8 + wave) * 16;
    const int laneHi = lane >> 4;
    const int ln     = lane & 15;
    const int sel    = laneHi ? 8 : 0;
    const int b      = m_base / T;                 // T % 16 == 0 -> tile within one batch row
    const float* uhrow = Uh + (size_t)b * HDIM;

    // A strip (16 rows x full K) in registers.
    const __bf16* arow = A + (size_t)(m_base + ln) * HDIM;
    bf16x16 a[16];
#pragma unroll
    for (int kk = 0; kk < 16; ++kk) a[kk] = load_op(arow, kk * 32, sel);

    // Prime the pipeline with B tile 0.
    async_b_tile(Bt, 0, &sB[0][0]);
    wait_async0();
    __syncthreads();

    float rs[8] = {0.f,0.f,0.f,0.f,0.f,0.f,0.f,0.f};
    int cur = 0;
    for (int nt = 0; nt < HDIM / 16; ++nt) {
        // Kick off the next tile's fill; it overlaps this tile's 16 WMMAs.
        if (nt + 1 < HDIM / 16) async_b_tile(Bt, (nt + 1) * 16, &sB[cur ^ 1][0]);

        const __bf16* brow = &sB[cur][0] + ln * BPAD;
        f32x8 acc = {};
#pragma unroll
        for (int kk = 0; kk < 16; ++kk) {
            bf16x16 bm = load_op_lds(brow, kk * 32, sel);
            acc = __builtin_amdgcn_wmma_f32_16x16x32_bf16(false, a[kk], false, bm,
                                                          (short)0, acc, false, false);
        }
        const int n = nt * 16 + ln;
        const float ub = uhrow[n] + biasN[n];
        const float vn = Vv[n];
        union { f32x8 v; float f[8]; } cu; cu.v = acc;
#pragma unroll
        for (int r = 0; r < 8; ++r) rs[r] += tanhf(cu.f[r] + ub) * vn;

        wait_async0();          // next tile landed in LDS
        __syncthreads();        // all waves done reading sB[cur], fill visible
        cur ^= 1;
    }

    // reduce over the 16 lanes of each half (xor offsets 1,2,4,8 stay within the half)
#pragma unroll
    for (int off = 1; off < 16; off <<= 1)
#pragma unroll
        for (int r = 0; r < 8; ++r) rs[r] += __shfl_xor(rs[r], off, 32);
    if (ln == 0) {
#pragma unroll
        for (int r = 0; r < 8; ++r) score[m_base + r + laneHi * 8] = rs[r];
    }
}

// ---------------- row softmax in place: s[b, 0..T) ----------------
__global__ void softmax_rows(float* __restrict__ s, int T) {
    __shared__ float red[8];
    float* row = s + (size_t)blockIdx.x * T;
    const int tid = threadIdx.x, lane = tid & 31, wave = tid >> 5;
    float mx = -1e30f;
    for (int t = tid; t < T; t += blockDim.x) mx = fmaxf(mx, row[t]);
#pragma unroll
    for (int off = 16; off >= 1; off >>= 1) mx = fmaxf(mx, __shfl_xor(mx, off, 32));
    if (lane == 0) red[wave] = mx;
    __syncthreads();
    mx = red[0];
#pragma unroll
    for (int w = 1; w < 8; ++w) mx = fmaxf(mx, red[w]);
    __syncthreads();
    float sum = 0.f;
    for (int t = tid; t < T; t += blockDim.x) { float e = __expf(row[t] - mx); row[t] = e; sum += e; }
#pragma unroll
    for (int off = 16; off >= 1; off >>= 1) sum += __shfl_xor(sum, off, 32);
    if (lane == 0) red[wave] = sum;
    __syncthreads();
    sum = 0.f;
#pragma unroll
    for (int w = 0; w < 8; ++w) sum += red[w];
    const float inv = 1.0f / sum;
    for (int t = tid; t < T; t += blockDim.x) row[t] *= inv;
}

// ---------------- out[b,h] = sum_t a[b,t] * X[b,t,h] ----------------
__global__ void weighted_sum(const float* __restrict__ a, const float* __restrict__ X,
                             float* __restrict__ out, int T) {
    const int b = blockIdx.x;
    const int h0 = threadIdx.x;                    // 256 threads, 2 columns each
    const float* Xb = X + (size_t)b * T * HDIM;
    const float* ab = a + (size_t)b * T;
    float s0 = 0.f, s1 = 0.f;
    for (int t = 0; t < T; ++t) {
        const float w = ab[t];
        s0 += w * Xb[(size_t)t * HDIM + h0];
        s1 += w * Xb[(size_t)t * HDIM + h0 + 256];
    }
    out[(size_t)b * HDIM + h0]       = s0;
    out[(size_t)b * HDIM + h0 + 256] = s1;
}

// ---------------- beta = softmax(tanh(concat(mv1,mt1)) @ wb)[0:2] ----------------
__global__ void beta_kernel(const float* __restrict__ mv1, const float* __restrict__ mt1,
                            const float* __restrict__ wb, float* __restrict__ beta) {
    __shared__ float z[2 * BDIM];
    const int j = threadIdx.x;                     // 128 threads
    const float* row = (j < BDIM) ? (mv1 + (size_t)j * HDIM) : (mt1 + (size_t)(j - BDIM) * HDIM);
    float s = 0.f;
    for (int h = 0; h < HDIM; ++h) s += tanhf(row[h]) * wb[h];
    z[j] = s;
    __syncthreads();
    if (j == 0) {
        float mx = z[0];
        for (int i = 1; i < 2 * BDIM; ++i) mx = fmaxf(mx, z[i]);
        float sum = 0.f;
        for (int i = 0; i < 2 * BDIM; ++i) sum += __expf(z[i] - mx);
        beta[0] = __expf(z[0] - mx) / sum;
        beta[1] = __expf(z[1] - mx) / sum;
    }
}

// ---------------- out = tanh(hWhh + b0*hv2 + b1*ht2 + bh) ----------------
__global__ void final_out(const float* __restrict__ hWhh, const float* __restrict__ hv2,
                          const float* __restrict__ ht2, const float* __restrict__ bh,
                          const float* __restrict__ beta, float* __restrict__ out) {
    const int i = blockIdx.x * blockDim.x + threadIdx.x;   // B*H
    const int n = i & (HDIM - 1);
    const float b0 = beta[0], b1 = beta[1];
    out[i] = tanhf(hWhh[i] + b0 * hv2[i] + b1 * ht2[i] + bh[n]);
}

extern "C" void kernel_launch(void* const* d_in, const int* in_sizes, int n_in,
                              void* d_out, int out_size, void* d_ws, size_t ws_size,
                              hipStream_t stream) {
    const float* h      = (const float*)d_in[0];
    const float* frames = (const float*)d_in[1];
    const float* text   = (const float*)d_in[2];
    const float* Wav = (const float*)d_in[3],  *Wat = (const float*)d_in[4];
    const float* Uav = (const float*)d_in[5],  *Uat = (const float*)d_in[6];
    const float* Vav = (const float*)d_in[7],  *Vat = (const float*)d_in[8];
    const float* bav = (const float*)d_in[9],  *bat = (const float*)d_in[10];
    const float* Whh = (const float*)d_in[11], *bh  = (const float*)d_in[12];
    const float* Wve = (const float*)d_in[13], *bve = (const float*)d_in[14];
    const float* Wqe = (const float*)d_in[15], *bqe = (const float*)d_in[16];
    const float* Wb  = (const float*)d_in[17];
    const float* Vbv = (const float*)d_in[18], *Vbt = (const float*)d_in[19];
    const float* bbv = (const float*)d_in[20], *bbt = (const float*)d_in[21];
    const float* wb  = (const float*)d_in[22];
    float* out = (float*)d_out;

    char* ws = (char*)d_ws;
    size_t off = 0;
    auto alloc = [&](size_t bytes) -> void* {
        off = (off + 255) & ~(size_t)255;
        void* p = ws + off;
        off += bytes;
        return p;
    };
    const size_t BH = (size_t)BDIM * HDIM, HH = (size_t)HDIM * HDIM;
    __bf16* h_bf   = (__bf16*)alloc(BH * 2);
    __bf16* fr_bf  = (__bf16*)alloc((size_t)BDIM * TV * HDIM * 2);
    __bf16* tx_bf  = (__bf16*)alloc((size_t)BDIM * TT * HDIM * 2);
    __bf16* WavT   = (__bf16*)alloc(HH * 2);
    __bf16* WatT   = (__bf16*)alloc(HH * 2);
    __bf16* UavT   = (__bf16*)alloc(HH * 2);
    __bf16* UatT   = (__bf16*)alloc(HH * 2);
    __bf16* WhhT   = (__bf16*)alloc(HH * 2);
    __bf16* WbT    = (__bf16*)alloc(HH * 2);
    __bf16* VbvT   = (__bf16*)alloc(HH * 2);
    __bf16* VbtT   = (__bf16*)alloc(HH * 2);
    __bf16* Wve_bf = (__bf16*)alloc(HH * 2);
    __bf16* Wqe_bf = (__bf16*)alloc(HH * 2);
    float* Uhv     = (float*)alloc(BH * 4);
    float* Uht     = (float*)alloc(BH * 4);
    float* score_v = (float*)alloc((size_t)BDIM * TV * 4);
    float* score_t = (float*)alloc((size_t)BDIM * TT * 4);
    float* hv_sum  = (float*)alloc(BH * 4);
    float* ht_sum  = (float*)alloc(BH * 4);
    __bf16* hv_bf  = (__bf16*)alloc(BH * 2);
    __bf16* ht_bf  = (__bf16*)alloc(BH * 2);
    float* hv2     = (float*)alloc(BH * 4);
    float* ht2     = (float*)alloc(BH * 4);
    float* Wbs     = (float*)alloc(BH * 4);
    float* mv1     = (float*)alloc(BH * 4);
    float* mt1     = (float*)alloc(BH * 4);
    float* hWhhBuf = (float*)alloc(BH * 4);
    float* beta    = (float*)alloc(256);

    // 1) conversions
    cvt_bf16<<<(int)((BH + 255) / 256), 256, 0, stream>>>(h, h_bf, (int)BH);
    {
        const int nf = BDIM * TV * HDIM;
        cvt_bf16<<<(nf + 255) / 256, 256, 0, stream>>>(frames, fr_bf, nf);
        const int nt = BDIM * TT * HDIM;
        cvt_bf16<<<(nt + 255) / 256, 256, 0, stream>>>(text, tx_bf, nt);
    }
    {
        const float* Wsrc[8] = {Wav, Wat, Uav, Uat, Whh, Wb, Vbv, Vbt};
        __bf16*      Wdst[8] = {WavT, WatT, UavT, UatT, WhhT, WbT, VbvT, VbtT};
        for (int i = 0; i < 8; ++i)
            cvt_tr_bf16<<<dim3(HDIM / 16, HDIM / 16), dim3(16, 16), 0, stream>>>(Wsrc[i], Wdst[i]);
        cvt_bf16<<<(int)((HH + 255) / 256), 256, 0, stream>>>(Wve, Wve_bf, (int)HH);
        cvt_bf16<<<(int)((HH + 255) / 256), 256, 0, stream>>>(Wqe, Wqe_bf, (int)HH);
    }

    // 2) small GEMMs from h  (M=64 -> 4 tiles -> 1 block of 8 waves)
    gemm_wmma<<<1, 256, 0, stream>>>(h_bf, UavT, Uhv, nullptr, nullptr, BDIM);
    gemm_wmma<<<1, 256, 0, stream>>>(h_bf, UatT, Uht, nullptr, nullptr, BDIM);
    gemm_wmma<<<1, 256, 0, stream>>>(h_bf, WbT,  Wbs, nullptr, nullptr, BDIM);
    gemm_wmma<<<1, 256, 0, stream>>>(h_bf, WhhT, hWhhBuf, nullptr, nullptr, BDIM);

    // 3) fused score GEMMs (dominant work: 32768x512x512 bf16 WMMA, async B tiles)
    gemm_score<<<(BDIM * TV / 16) / 8, 256, 0, stream>>>(fr_bf, WavT, Uhv, bav, Vav,
                                                         score_v, TV);
    gemm_score<<<(BDIM * TT / 16) / 8, 256, 0, stream>>>(tx_bf, WatT, Uht, bat, Vat,
                                                         score_t, TT);

    // 4) softmax over T (in place)
    softmax_rows<<<BDIM, 256, 0, stream>>>(score_v, TV);
    softmax_rows<<<BDIM, 256, 0, stream>>>(score_t, TT);

    // 5) attention-weighted sums (fp32 inputs, coalesced over H)
    weighted_sum<<<BDIM, 256, 0, stream>>>(score_v, frames, hv_sum, TV);
    weighted_sum<<<BDIM, 256, 0, stream>>>(score_t, text, ht_sum, TT);

    cvt_bf16<<<(int)((BH + 255) / 256), 256, 0, stream>>>(hv_sum, hv_bf, (int)BH);
    cvt_bf16<<<(int)((BH + 255) / 256), 256, 0, stream>>>(ht_sum, ht_bf, (int)BH);

    // 6) encoders + gating pre-activations
    gemm_wmma<<<1, 256, 0, stream>>>(hv_bf, Wve_bf, hv2, bve, nullptr, BDIM);  // hv_sum @ Wve.T + bve
    gemm_wmma<<<1, 256, 0, stream>>>(ht_bf, Wqe_bf, ht2, bqe, nullptr, BDIM);  // ht_sum @ Wqe.T + bqe
    gemm_wmma<<<1, 256, 0, stream>>>(hv_bf, VbvT, mv1, bbv, Wbs, BDIM);        // hv_sum @ Vbv + bbv + Wbs
    gemm_wmma<<<1, 256, 0, stream>>>(ht_bf, VbtT, mt1, bbt, Wbs, BDIM);        // ht_sum @ Vbt + bbt + Wbs

    // 7) beta and final combine
    beta_kernel<<<1, 2 * BDIM, 0, stream>>>(mv1, mt1, wb, beta);
    final_out<<<(int)(BH / 256), 256, 0, stream>>>(hWhhBuf, hv2, ht2, bh, beta, out);
}
